// AR_decoder_82592221102346
// MI455X (gfx1250) — matrix operand order repaired and checked by reading, not compiled, not run on hardware
//
#include <hip/hip_runtime.h>

typedef __attribute__((ext_vector_type(16))) _Float16 v16h;
typedef __attribute__((ext_vector_type(8)))  float    v8f;

// Problem constants (IN_SIZE == HIDDEN == 64, N_CLASSES == 5)
constexpr int kIn = 64;
constexpr int kH  = 64;
constexpr int kNC = 5;
constexpr int kGP = 20;   // padded gates row stride (floats) to spread LDS banks

__device__ __forceinline__ v8f wmma_f16(v16h a, v16h b, v8f c) {
    return __builtin_amdgcn_wmma_f32_16x16x32_f16(
        /*neg_a=*/false, a, /*neg_b=*/false, b,
        /*c_mod=*/(short)0, c, /*reuse_a=*/false, /*reuse_b=*/false);
}

__device__ __forceinline__ float sigf(float x) {
    x = fminf(fmaxf(x, -30.f), 30.f);
    return 1.0f / (1.0f + __expf(-x));
}
__device__ __forceinline__ float tanh_fast(float x) {
    x = fminf(fmaxf(x, -15.f), 15.f);
    float e = __expf(-2.0f * x);
    return (1.0f - e) / (1.0f + e);
}

// Load one time step of x (row m = lane%16) into A-fragment staging order.
// A-frag halfs: j<8  -> K = 32c + 8*lhalf + j
//               j>=8 -> K = 32c + 16 + 8*lhalf + (j-8)     (ISA 7.12.2)
__device__ __forceinline__ void load_x_stage(const float* xp, int lhalf, float* st) {
#pragma unroll
    for (int c = 0; c < 2; ++c) {
        const float4* p0 = (const float4*)(xp + 32 * c + 8 * lhalf);
        const float4* p1 = (const float4*)(xp + 32 * c + 16 + 8 * lhalf);
        float4 a0 = p0[0], a1 = p0[1], q0 = p1[0], q1 = p1[1];
        st[16*c+ 0]=a0.x; st[16*c+ 1]=a0.y; st[16*c+ 2]=a0.z; st[16*c+ 3]=a0.w;
        st[16*c+ 4]=a1.x; st[16*c+ 5]=a1.y; st[16*c+ 6]=a1.z; st[16*c+ 7]=a1.w;
        st[16*c+ 8]=q0.x; st[16*c+ 9]=q0.y; st[16*c+10]=q0.z; st[16*c+11]=q0.w;
        st[16*c+12]=q1.x; st[16*c+13]=q1.y; st[16*c+14]=q1.z; st[16*c+15]=q1.w;
    }
}

__launch_bounds__(256, 1)
__global__ void ar_lstm_decoder_kernel(const float* __restrict__ x,
                                       const float* __restrict__ W_ih,  // [256,128]
                                       const float* __restrict__ W_hh,  // [256,64]
                                       const float* __restrict__ b_ih,  // [256]
                                       const float* __restrict__ b_hh,  // [256]
                                       const float* __restrict__ W_fc,  // [5,64]
                                       const float* __restrict__ b_fc,  // [5]
                                       const float* __restrict__ emb,   // [5,64]
                                       float* __restrict__ out,         // [B,T,5]
                                       int T)
{
    // LDS: h and prev stored directly in WMMA A-fragment layout (f16),
    // gates staged fp32 with padded stride, logits tiny.
    __shared__ __align__(16) _Float16 hA[2][32][16];   // [Kchunk][lane][half]
    __shared__ __align__(16) _Float16 pA[2][32][16];
    __shared__ __align__(16) float    gatesS[256 * kGP]; // [N=256][m padded]
    __shared__ float logitsS[16][8];

    const int tid   = threadIdx.x;
    const int wave  = tid >> 5;
    const int lane  = tid & 31;
    const int lhalf = lane >> 4;     // 0: lanes 0-15, 1: lanes 16-31
    const int lm    = lane & 15;
    const int b0    = blockIdx.x * 16;

    // ---------------- one-time setup: weights -> f16 B-fragments in VGPRs ---
    // Wave w owns gate N-tiles {2w, 2w+1}; tile n covers gate rows 16n..16n+15.
    // B-frag halfs: j<8 -> K = 8*lhalf + j, j>=8 -> K = 16 + 8*lhalf + (j-8) (within 32-wide K chunk), N = lm.
    v16h Bfrag[2][6];
    float biasv[2];
#pragma unroll
    for (int i = 0; i < 2; ++i) {
        const int n  = 2 * wave + i;
        const int Ng = 16 * n + lm;               // gate row 0..255
        biasv[i] = b_ih[Ng] + b_hh[Ng];
#pragma unroll
        for (int c = 0; c < 6; ++c) {
            const float* Wrow = (c < 4) ? (W_ih + (size_t)Ng * 128)
                                        : (W_hh + (size_t)Ng * 64);
            const int kb = ((c < 4) ? 32 * c : 32 * (c - 4)) + 8 * lhalf;
            v16h f;
#pragma unroll
            for (int j = 0; j < 16; ++j) f[j] = (_Float16)Wrow[kb + (j < 8 ? j : j + 8)];
            Bfrag[i][c] = f;
        }
    }
    // Wave 0 additionally keeps W_fc^T as B-fragments (N padded 5->16 with 0).
    v16h Bfc[2];
#pragma unroll
    for (int c = 0; c < 2; ++c) {
        v16h f;
#pragma unroll
        for (int j = 0; j < 16; ++j) {
            const int K = 32 * c + 8 * lhalf + (j < 8 ? j : j + 8);
            f[j] = (lm < kNC) ? (_Float16)W_fc[lm * kH + K] : (_Float16)0.f;
        }
        Bfc[c] = f;
    }

    // zero initial state (h = c = prev = 0)
    for (int i = tid; i < 2 * 32 * 16; i += 256) {
        ((_Float16*)hA)[i] = (_Float16)0.f;
        ((_Float16*)pA)[i] = (_Float16)0.f;
    }
    // cell-state ownership: thread owns rows mbase..mbase+3 at column ccol
    const int ccol  = tid & 63;
    const int mbase = (tid >> 6) << 2;
    float creg[4] = {0.f, 0.f, 0.f, 0.f};

    const float* xrow = x + ((size_t)(b0 + lm)) * (size_t)T * kIn;

    // software pipeline: stage x(t=0)
    float xstage[32];
    load_x_stage(xrow, lhalf, xstage);
    __syncthreads();

#pragma unroll 1
    for (int t = 0; t < T; ++t) {
        // ---- convert staged x to f16 A-fragments; prefetch next step -------
        v16h Ax[2];
#pragma unroll
        for (int c = 0; c < 2; ++c) {
            v16h f;
#pragma unroll
            for (int j = 0; j < 16; ++j) f[j] = (_Float16)xstage[16 * c + j];
            Ax[c] = f;
        }
        if (t + 1 < T) load_x_stage(xrow + (size_t)(t + 1) * kIn, lhalf, xstage);
        if (t + 8 < T) __builtin_prefetch(xrow + (size_t)(t + 8) * kIn, 0, 0);

        v16h Ap[2], Ah[2];
#pragma unroll
        for (int c = 0; c < 2; ++c) {
            Ap[c] = *(const v16h*)&pA[c][lane][0];
            Ah[c] = *(const v16h*)&hA[c][lane][0];
        }

        // ---- phase 1: gates = [x|prev]@W_ih^T + h@W_hh^T + bias (WMMA) -----
#pragma unroll
        for (int i = 0; i < 2; ++i) {
            v8f acc;
#pragma unroll
            for (int j = 0; j < 8; ++j) acc[j] = biasv[i];
            acc = wmma_f16(Ax[0], Bfrag[i][0], acc);
            acc = wmma_f16(Ax[1], Bfrag[i][1], acc);
            acc = wmma_f16(Ap[0], Bfrag[i][2], acc);
            acc = wmma_f16(Ap[1], Bfrag[i][3], acc);
            acc = wmma_f16(Ah[0], Bfrag[i][4], acc);
            acc = wmma_f16(Ah[1], Bfrag[i][5], acc);
            // C layout: lane holds N = 16n+lm, M = vgpr + 8*lhalf
            float* gp = &gatesS[(16 * (2 * wave + i) + lm) * kGP + 8 * lhalf];
            *(float4*)gp       = make_float4(acc[0], acc[1], acc[2], acc[3]);
            *((float4*)gp + 1) = make_float4(acc[4], acc[5], acc[6], acc[7]);
        }
        __syncthreads();   // gates ready

        // ---- phase 2: LSTM cell update, h -> LDS A-fragments (f16) --------
        {
            const float4 gi = *(const float4*)&gatesS[(ccol      ) * kGP + mbase];
            const float4 gf = *(const float4*)&gatesS[(ccol +  64) * kGP + mbase];
            const float4 gg = *(const float4*)&gatesS[(ccol + 128) * kGP + mbase];
            const float4 go = *(const float4*)&gatesS[(ccol + 192) * kGP + mbase];
            const float iv[4] = {gi.x, gi.y, gi.z, gi.w};
            const float fv[4] = {gf.x, gf.y, gf.z, gf.w};
            const float gv[4] = {gg.x, gg.y, gg.z, gg.w};
            const float ov[4] = {go.x, go.y, go.z, go.w};
            const int kc = ccol & 31, chunk = ccol >> 5, grp = kc >> 3;
            const int lrow = 16 * (grp & 1);
            const int hidx = 8 * (grp >> 1) + (kc & 7);
#pragma unroll
            for (int j = 0; j < 4; ++j) {
                const float cn = sigf(fv[j]) * creg[j] + sigf(iv[j]) * tanh_fast(gv[j]);
                creg[j] = cn;
                const float hv = sigf(ov[j]) * tanh_fast(cn);
                hA[chunk][(mbase + j) + lrow][hidx] = (_Float16)hv;
            }
        }
        __syncthreads();   // h ready

        // ---- phase 3: FC logits via WMMA on wave 0 ------------------------
        if (wave == 0) {
            v8f acc = {};
            acc = wmma_f16(*(const v16h*)&hA[0][lane][0], Bfc[0], acc);
            acc = wmma_f16(*(const v16h*)&hA[1][lane][0], Bfc[1], acc);
            if (lm < kNC) {
#pragma unroll
                for (int v = 0; v < 8; ++v) logitsS[8 * lhalf + v][lm] = acc[v];
            }
        }
        __syncthreads();   // logits ready

        // ---- phase 4: log_softmax + argmax + embedding feedback -----------
        if (tid < 16) {
            const int m = tid;
            float l[kNC], mx = -1e30f;
#pragma unroll
            for (int c = 0; c < kNC; ++c) {
                l[c] = logitsS[m][c] + b_fc[c];
                mx = fmaxf(mx, l[c]);
            }
            float s = 0.f;
#pragma unroll
            for (int c = 0; c < kNC; ++c) s += __expf(l[c] - mx);
            const float lse = mx + __logf(s);
            int best = 0; float bv = l[0];
            float* op = out + (((size_t)(b0 + m)) * (size_t)T + t) * kNC;
#pragma unroll
            for (int c = 0; c < kNC; ++c) {
                op[c] = l[c] - lse;
                if (l[c] > bv) { bv = l[c]; best = c; }
            }
            // prev = emb[best] -> pA in A-fragment layout (pairs -> b32 store)
            const float* e = emb + best * kIn;
#pragma unroll
            for (int col = 0; col < kIn; col += 2) {
                const int kc = col & 31, chunk = col >> 5, grp = kc >> 3;
                const int lrow = m + 16 * (grp & 1);
                const int hidx = 8 * (grp >> 1) + (kc & 7);
                pA[chunk][lrow][hidx]     = (_Float16)e[col];
                pA[chunk][lrow][hidx + 1] = (_Float16)e[col + 1];
            }
        }
        __syncthreads();   // prev ready for next step
    }
}

extern "C" void kernel_launch(void* const* d_in, const int* in_sizes, int n_in,
                              void* d_out, int out_size, void* d_ws, size_t ws_size,
                              hipStream_t stream) {
    (void)n_in; (void)d_ws; (void)ws_size; (void)out_size;
    const float* x     = (const float*)d_in[0];
    // d_in[1] x_lengths (int), d_in[2] edge_list (int): unused by reference math
    const float* W_ih  = (const float*)d_in[3];
    const float* W_hh  = (const float*)d_in[4];
    const float* b_ih  = (const float*)d_in[5];
    const float* b_hh  = (const float*)d_in[6];
    const float* W_fc  = (const float*)d_in[7];
    const float* b_fc  = (const float*)d_in[8];
    const float* emb   = (const float*)d_in[9];
    float* out = (float*)d_out;

    const int B = in_sizes[1];                       // x_lengths has B entries
    const int T = in_sizes[0] / (B * kIn);           // x is [B,T,64]

    dim3 grid(B / 16), block(256);
    ar_lstm_decoder_kernel<<<grid, block, 0, stream>>>(
        x, W_ih, W_hh, b_ih, b_hh, W_fc, b_fc, emb, out, T);
}
